// MoleculeEmbeddingNetwork_62981400429191
// MI455X (gfx1250) — matrix operand order repaired and checked by reading, not compiled, not run on hardware
//
#include <hip/hip_runtime.h>
#include <hip/hip_bf16.h>
#include <math.h>

// ---------------------------------------------------------------------------
// Types for CDNA5 WMMA (wave32):  D(16x16 f32) = A(16x32 bf16) x B(32x16 bf16) + C
// ---------------------------------------------------------------------------
typedef __bf16 bf16_t;
typedef __attribute__((ext_vector_type(16))) __bf16 v16bf;
typedef __attribute__((ext_vector_type(8)))  float  v8f;

// A fragment from a row-major bf16 tile [16][ldk] in LDS.
// ISA 16-bit A layout: lanes 0-15 hold M=0..15 (k = k0+0..7, k0+16..23),
// lanes 16-31 hold M=0..15 (k = k0+8..15, k0+24..31).
static __device__ __forceinline__ v16bf load_a_frag(const bf16_t* A, int ldk, int k0, int lane) {
  const int m    = lane & 15;
  const int half = lane >> 4;
  const bf16_t* p0 = A + m * ldk + (k0 + half * 8);
  const bf16_t* p1 = p0 + 16;
  v16bf r;
#pragma unroll
  for (int i = 0; i < 8; ++i) { r[i] = p0[i]; r[i + 8] = p1[i]; }
  return r;
}

// B fragment from a transposed weight matrix WT[n][k] (row-major, ldk = padded K).
// ISA 16-bit B layout: lane = column n (L&15); lanes 0-15 hold K=k0+0..7, k0+16..23,
// lanes 16-31 hold K=k0+8..15, k0+24..31 (same k order as A).
static __device__ __forceinline__ v16bf load_b_frag(const bf16_t* WT, int ldk, int ntile, int k0, int lane) {
  const int n    = ntile * 16 + (lane & 15);
  const int half = lane >> 4;
  const bf16_t* p = WT + (size_t)n * ldk + (k0 + half * 8);
  v16bf r;
#pragma unroll
  for (int i = 0; i < 8; ++i) { r[i] = p[i]; r[i + 8] = p[i + 16]; }
  return r;
}

static __device__ __forceinline__ v8f wmma_bf16(v16bf a, v16bf b, v8f c) {
  return __builtin_amdgcn_wmma_f32_16x16x32_bf16(false, a, false, b, (short)0, c, false, false);
}

static __device__ __forceinline__ v8f v8f_zero() {
  v8f z = {0.f, 0.f, 0.f, 0.f, 0.f, 0.f, 0.f, 0.f};
  return z;
}

// ---------------------------------------------------------------------------
// Weight convert: W[K][N] f32 -> WT[N][Kpad] bf16 (zero-padded)
// ---------------------------------------------------------------------------
__global__ void wconv_kernel(const float* __restrict__ W, bf16_t* __restrict__ WT,
                             int K, int Ncols, int Kpad) {
  int total = Ncols * Kpad;
  for (int i = blockIdx.x * blockDim.x + threadIdx.x; i < total; i += gridDim.x * blockDim.x) {
    int n = i / Kpad, k = i % Kpad;
    float v = (k < K) ? W[(size_t)k * Ncols + n] : 0.0f;
    WT[i] = (bf16_t)v;
  }
}

// ---------------------------------------------------------------------------
// Fused atom MLP: relu(feat@Wf+bf) -> concat(msg) -> relu(@W1+b1) -> tanh(@W2+b2)
// One wave per 16-row tile; 4 waves (64 rows) per block. Wave-private LDS strip.
// ---------------------------------------------------------------------------
#define ATOM_KAD 384

__global__ __launch_bounds__(128) void atom_kernel(
    const float* __restrict__ feat, const float* __restrict__ msg,
    const bf16_t* __restrict__ WfT, const float* __restrict__ bfeat,
    const bf16_t* __restrict__ W1T, const float* __restrict__ b1,
    const bf16_t* __restrict__ W2T, const float* __restrict__ b2,
    float* __restrict__ atom_out, int N) {
  __shared__ bf16_t sAD[4][16][ATOM_KAD];  // 48 KB
  const int wave = threadIdx.x >> 5, lane = threadIdx.x & 31;
  const int rowbase = blockIdx.x * 64 + wave * 16;
  bf16_t(*A)[ATOM_KAD] = sAD[wave];
  const int m = lane & 15, half = lane >> 4;

  // Stage atom_feature -> cols [0,64), zero-padded K 39->64
  for (int i = lane; i < 16 * 64; i += 32) {
    int r = i >> 6, c = i & 63;
    int row = rowbase + r;
    float v = (c < 39 && row < N) ? feat[(size_t)row * 39 + c] : 0.0f;
    A[r][c] = (bf16_t)v;
  }
  // Stage atom_message -> cols [128,384)
  for (int i = lane; i < 16 * 256; i += 32) {
    int r = i >> 8, c = i & 255;
    int row = rowbase + r;
    float v = (row < N) ? msg[(size_t)row * 256 + c] : 0.0f;
    A[r][128 + c] = (bf16_t)v;
  }

  // ---- GEMM1: [16x64] x WfT -> feat_emb [16x128], relu; overwrite cols [0,128)
  {
    v8f acc[8];
#pragma unroll
    for (int t = 0; t < 8; ++t) acc[t] = v8f_zero();
    for (int kb = 0; kb < 2; ++kb) {
      v16bf a = load_a_frag(&A[0][0], ATOM_KAD, kb * 32, lane);
#pragma unroll
      for (int t = 0; t < 8; ++t) {
        v16bf b = load_b_frag(WfT, 64, t, kb * 32, lane);
        acc[t] = wmma_bf16(a, b, acc[t]);
      }
    }
#pragma unroll
    for (int t = 0; t < 8; ++t) {
      float bias = bfeat[t * 16 + m];
#pragma unroll
      for (int v = 0; v < 8; ++v) {
        float x = acc[t][v] + bias;
        A[v + half * 8][t * 16 + m] = (bf16_t)(x > 0.0f ? x : 0.0f);
      }
    }
  }

  // ---- GEMM2: [16x384] x W1T -> h [16x128], relu; overwrite cols [0,128)
  {
    v8f acc[8];
#pragma unroll
    for (int t = 0; t < 8; ++t) acc[t] = v8f_zero();
    for (int kb = 0; kb < 12; ++kb) {
      v16bf a = load_a_frag(&A[0][0], ATOM_KAD, kb * 32, lane);
#pragma unroll
      for (int t = 0; t < 8; ++t) {
        v16bf b = load_b_frag(W1T, 384, t, kb * 32, lane);
        acc[t] = wmma_bf16(a, b, acc[t]);
      }
    }
#pragma unroll
    for (int t = 0; t < 8; ++t) {
      float bias = b1[t * 16 + m];
#pragma unroll
      for (int v = 0; v < 8; ++v) {
        float x = acc[t][v] + bias;
        A[v + half * 8][t * 16 + m] = (bf16_t)(x > 0.0f ? x : 0.0f);
      }
    }
  }

  // ---- GEMM3: [16x128] x W2T -> tanh -> atom_out f32
  {
    v8f acc[8];
#pragma unroll
    for (int t = 0; t < 8; ++t) acc[t] = v8f_zero();
    for (int kb = 0; kb < 4; ++kb) {
      v16bf a = load_a_frag(&A[0][0], ATOM_KAD, kb * 32, lane);
#pragma unroll
      for (int t = 0; t < 8; ++t) {
        v16bf b = load_b_frag(W2T, 128, t, kb * 32, lane);
        acc[t] = wmma_bf16(a, b, acc[t]);
      }
    }
#pragma unroll
    for (int t = 0; t < 8; ++t) {
      float bias = b2[t * 16 + m];
#pragma unroll
      for (int v = 0; v < 8; ++v) {
        int row = rowbase + v + half * 8;
        if (row < N) atom_out[(size_t)row * 128 + t * 16 + m] = tanhf(acc[t][v] + bias);
      }
    }
  }
}

// ---------------------------------------------------------------------------
// Gather-GEMM: out[r] = tanh(concat(feat[idx[r]], msg[idx[r]]) @ WT + bias)
// K = 295 padded to 320, Nout = 256 (two halves of 8 n-tiles to limit VGPRs).
// base_rows_dev: optional device scalar; output shifted by (*base_rows_dev) rows.
// ---------------------------------------------------------------------------
#define G_K 320

__global__ __launch_bounds__(128) void gather_gemm_kernel(
    const float* __restrict__ feat, const float* __restrict__ msg,
    const int* __restrict__ idx, int R,
    const bf16_t* __restrict__ WT, const float* __restrict__ bias,
    float* __restrict__ out, const int* __restrict__ base_rows_dev) {
  __shared__ bf16_t sA[4][16][G_K];  // 40 KB
  const int wave = threadIdx.x >> 5, lane = threadIdx.x & 31;
  const int rowbase = blockIdx.x * 64 + wave * 16;
  bf16_t(*A)[G_K] = sA[wave];
  const int m = lane & 15, half = lane >> 4;

  // Stage gathered amf rows: [0,39)=feat, [39,295)=msg, [295,320)=0
  for (int i = lane; i < 16 * G_K; i += 32) {
    int r = i / G_K, c = i % G_K;
    int er = rowbase + r;
    float v = 0.0f;
    if (er < R) {
      int g = idx[er];
      if (c < 39)       v = feat[(size_t)g * 39 + c];
      else if (c < 295) v = msg[(size_t)g * 256 + (c - 39)];
    }
    A[r][c] = (bf16_t)v;
  }

  size_t obase = base_rows_dev ? (size_t)base_rows_dev[0] * 256 : 0;

  for (int nh = 0; nh < 2; ++nh) {
    v8f acc[8];
#pragma unroll
    for (int t = 0; t < 8; ++t) acc[t] = v8f_zero();
    for (int kb = 0; kb < 10; ++kb) {
      v16bf a = load_a_frag(&A[0][0], G_K, kb * 32, lane);
#pragma unroll
      for (int t = 0; t < 8; ++t) {
        v16bf b = load_b_frag(WT, G_K, nh * 8 + t, kb * 32, lane);
        acc[t] = wmma_bf16(a, b, acc[t]);
      }
    }
#pragma unroll
    for (int t = 0; t < 8; ++t) {
      int col = (nh * 8 + t) * 16 + m;
      float bi = bias[col];
#pragma unroll
      for (int v = 0; v < 8; ++v) {
        int er = rowbase + v + half * 8;
        if (er < R) out[obase + (size_t)er * 256 + col] = tanhf(acc[t][v] + bi);
      }
    }
  }
}

// ---------------------------------------------------------------------------
// Segment pooling via binary search over sorted segment ids (deterministic).
// ---------------------------------------------------------------------------
static __device__ __forceinline__ int lower_bound_i(const int* a, int n, int key) {
  int lo = 0, hi = n;
  while (lo < hi) {
    int mid = (lo + hi) >> 1;
    if (a[mid] < key) lo = mid + 1; else hi = mid;
  }
  return lo;
}

__global__ __launch_bounds__(128) void mol_pool_kernel(
    const float* __restrict__ atom_out, const int* __restrict__ seg, int N,
    const int* __restrict__ n_mols_dev, float* __restrict__ out) {
  int M = n_mols_dev[0];
  int mol = blockIdx.x;
  if (mol >= M) return;
  int col = threadIdx.x;  // 128 columns
  int lo = lower_bound_i(seg, N, mol);
  int hi = lower_bound_i(seg, N, mol + 1);
  float s = 0.0f, mx = -3.402823466e38f;
  for (int i = lo; i < hi; ++i) {
    float v = atom_out[(size_t)i * 128 + col];
    s += v;
    mx = fmaxf(mx, v);
  }
  int cnt = hi - lo;
  float avg = s / fmaxf((float)cnt, 1.0f);
  out[(size_t)mol * 256 + col] = avg;                       // avg_pool half
  out[(size_t)mol * 256 + 128 + col] = cnt > 0 ? mx : 0.0f; // max_pool half
}

__global__ __launch_bounds__(256) void ring_pool_kernel(
    const float* __restrict__ entries, const int* __restrict__ seg, int E,
    const int* __restrict__ n_rings_dev, const int* __restrict__ n_mols_dev,
    int NL, float* __restrict__ out) {
  int NR = n_rings_dev[0];
  int ring = blockIdx.x;
  if (ring >= NR) return;
  int col = threadIdx.x;  // 256 columns
  int lo = lower_bound_i(seg, E, ring);
  int hi = lower_bound_i(seg, E, ring + 1);
  float s = 0.0f;
  for (int i = lo; i < hi; ++i) s += entries[(size_t)i * 256 + col];
  size_t base = ((size_t)n_mols_dev[0] + (size_t)NL) * 256;
  out[base + (size_t)ring * 256 + col] = s;
}

// ---------------------------------------------------------------------------
// Host launcher
// ---------------------------------------------------------------------------
extern "C" void kernel_launch(void* const* d_in, const int* in_sizes, int n_in,
                              void* d_out, int out_size, void* d_ws, size_t ws_size,
                              hipStream_t stream) {
  (void)n_in; (void)ws_size;
  const float* atom_feature = (const float*)d_in[0];
  const float* atom_message = (const float*)d_in[1];
  const float* W_feat = (const float*)d_in[2];
  const float* b_feat = (const float*)d_in[3];
  const float* W_d1   = (const float*)d_in[4];
  const float* b_d1   = (const float*)d_in[5];
  const float* W_d2   = (const float*)d_in[6];
  const float* b_d2   = (const float*)d_in[7];
  const float* W_leaf = (const float*)d_in[8];
  const float* b_leaf = (const float*)d_in[9];
  const float* W_ring = (const float*)d_in[10];
  const float* b_ring = (const float*)d_in[11];
  const int* atom_seg   = (const int*)d_in[12];
  const int* leaf_idx   = (const int*)d_in[13];
  const int* ring_idx   = (const int*)d_in[14];
  const int* ring_seg   = (const int*)d_in[15];
  const int* n_mols_dev  = (const int*)d_in[16];
  const int* n_rings_dev = (const int*)d_in[17];

  const int N   = in_sizes[0] / 39;
  const int NL  = in_sizes[13];
  const int NRE = in_sizes[14];

  // Workspace carving (256B aligned)
  size_t off = 0;
  auto carve = [&](size_t bytes) -> void* {
    off = (off + 255) & ~(size_t)255;
    void* p = (char*)d_ws + off;
    off += bytes;
    return p;
  };
  bf16_t* WfT = (bf16_t*)carve((size_t)128 * 64 * sizeof(bf16_t));
  bf16_t* W1T = (bf16_t*)carve((size_t)128 * 384 * sizeof(bf16_t));
  bf16_t* W2T = (bf16_t*)carve((size_t)128 * 128 * sizeof(bf16_t));
  bf16_t* WlT = (bf16_t*)carve((size_t)256 * 320 * sizeof(bf16_t));
  bf16_t* WrT = (bf16_t*)carve((size_t)256 * 320 * sizeof(bf16_t));
  float* atom_out     = (float*)carve((size_t)N * 128 * sizeof(float));
  float* ring_entries = (float*)carve((size_t)NRE * 256 * sizeof(float));

  // 1) Weight transpose + pad + bf16 convert
  wconv_kernel<<<32, 256, 0, stream>>>(W_feat, WfT, 39, 128, 64);
  wconv_kernel<<<64, 256, 0, stream>>>(W_d1, W1T, 384, 128, 384);
  wconv_kernel<<<32, 256, 0, stream>>>(W_d2, W2T, 128, 128, 128);
  wconv_kernel<<<128, 256, 0, stream>>>(W_leaf, WlT, 295, 256, 320);
  wconv_kernel<<<128, 256, 0, stream>>>(W_ring, WrT, 295, 256, 320);

  // 2) Fused atom MLP -> atom_out [N x 128]
  atom_kernel<<<(N + 63) / 64, 128, 0, stream>>>(
      atom_feature, atom_message, WfT, b_feat, W1T, b_d1, W2T, b_d2, atom_out, N);

  // 3) Leaf embeddings -> d_out rows [M, M+NL)   (base via device scalar)
  gather_gemm_kernel<<<(NL + 63) / 64, 128, 0, stream>>>(
      atom_feature, atom_message, leaf_idx, NL, WlT, b_leaf, (float*)d_out, n_mols_dev);

  // 4) Ring entry embeddings (gather commutes with tanh/GEMM) -> ws [NRE x 256]
  gather_gemm_kernel<<<(NRE + 63) / 64, 128, 0, stream>>>(
      atom_feature, atom_message, ring_idx, NRE, WrT, b_ring, ring_entries, nullptr);

  // 5) Pooling. Grid bound = M + NR (derived from out_size); blocks self-guard.
  int maxSeg = out_size / 256 - NL;  // == n_mols + n_rings
  if (maxSeg > 0) {
    mol_pool_kernel<<<maxSeg, 128, 0, stream>>>(atom_out, atom_seg, N, n_mols_dev, (float*)d_out);
    ring_pool_kernel<<<maxSeg, 256, 0, stream>>>(ring_entries, ring_seg, NRE,
                                                 n_rings_dev, n_mols_dev, NL, (float*)d_out);
  }
}
